// CausalAttention_77610059039141
// MI455X (gfx1250) — matrix operand order repaired and checked by reading, not compiled, not run on hardware
//
#include <hip/hip_runtime.h>
#include <math.h>

#define S_LEN 4096
#define F_DIM 128
#define H_NUM 4
#define D_DIM 32
#define B_NUM 4
#define M_ROWS (B_NUM * S_LEN)   // 16384
#define A_STRIDE 144             // bf16 row stride in LDS (pad: banks spread, 16B align)
#define Y_STRIDE 132             // f32 row stride for epilogue buffer
#define WMAT_ELEMS 16384         // 128x128 per plane
#define WMAT_STRIDE (2 * WMAT_ELEMS) // hi+lo planes per matrix (ushorts)

typedef __attribute__((ext_vector_type(16))) __bf16 v16bf;
typedef __attribute__((ext_vector_type(8)))  float  v8f;
typedef __attribute__((ext_vector_type(16))) unsigned short v16us;

__device__ __forceinline__ unsigned short f2bf(float f) {
  unsigned u = __float_as_uint(f);
  u += 0x7FFFu + ((u >> 16) & 1u);       // round-to-nearest-even
  return (unsigned short)(u >> 16);
}
__device__ __forceinline__ float bf2f(unsigned short h) {
  return __uint_as_float(((unsigned)h) << 16);
}

// A-operand permutation (CDNA5 ISA §7.12.2, 16-bit A 16x32):
// K = kc*32 + q ; h=(q>>3)&1 ; e=(q&7) + 8*((q>>4)&1) ; block = kc*2+h
__device__ __forceinline__ int aperm(int K) {
  const int kc = K >> 5, q = K & 31;
  const int h = (q >> 3) & 1;
  const int e = (q & 7) + (((q >> 4) & 1) << 3);
  return ((kc << 1) + h) * 16 + e;
}

// A 16x32 bf16 operand: single contiguous 32B per lane from permuted LDS plane
__device__ __forceinline__ v16bf load_a(const unsigned short* plane, int row, int kc, int lane) {
  const unsigned short* p = plane + row * A_STRIDE + (((kc << 1) + (lane >> 4)) << 4);
  return __builtin_bit_cast(v16bf, *(const v16us*)p);
}

// B 32x16 bf16 operand from pre-swizzled weight buffer: 32B coalesced per lane
struct BPair { v16bf hi, lo; };
__device__ __forceinline__ BPair load_w_tile(const unsigned short* __restrict__ Wz,
                                             int nt, int kc, int lane) {
  const int off = (((nt << 2) + kc) * 32 + lane) << 4;
  BPair r;
  r.hi = __builtin_bit_cast(v16bf, *(const v16us*)(Wz + off));
  r.lo = __builtin_bit_cast(v16bf, *(const v16us*)(Wz + WMAT_ELEMS + off));
  return r;
}

// bf16x3 split accumulation: A*B ~= Ah*Bl + Al*Bh + Ah*Bh (f32 accumulate)
__device__ __forceinline__ v8f mm_acc(v8f c, v16bf ah, v16bf al, v16bf bh, v16bf bl) {
  c = __builtin_amdgcn_wmma_f32_16x16x32_bf16(false, ah, false, bl, (short)0, c, false, false);
  c = __builtin_amdgcn_wmma_f32_16x16x32_bf16(false, al, false, bh, (short)0, c, false, false);
  c = __builtin_amdgcn_wmma_f32_16x16x32_bf16(false, ah, false, bh, (short)0, c, false, false);
  return c;
}

// full K=128 tile: preload all 4 B chunk-pairs (one clause, one wait), then 12 WMMAs
__device__ __forceinline__ v8f mm_tile(const unsigned short* __restrict__ Wz, int nt,
                                       const v16bf* Ah, const v16bf* Al, int lane) {
  BPair bp[4];
  #pragma unroll
  for (int kc = 0; kc < 4; ++kc) bp[kc] = load_w_tile(Wz, nt, kc, lane);
  v8f c = {};
  #pragma unroll
  for (int kc = 0; kc < 4; ++kc) c = mm_acc(c, Ah[kc], Al[kc], bp[kc].hi, bp[kc].lo);
  return c;
}

// stage a 128x128 f32 block -> permuted bf16 hi/lo planes in LDS
__device__ __forceinline__ void stage_block(const float* __restrict__ src, int rowBase,
                                            unsigned short* hi, unsigned short* lo, int tid) {
  #pragma unroll
  for (int it = 0; it < 16; ++it) {
    const int idx4 = tid + 256 * it;
    const int row = idx4 >> 5, c4 = (idx4 & 31) * 4;
    const float4 f = *(const float4*)(src + (rowBase + row) * F_DIM + c4);
    const int dbase = row * A_STRIDE + aperm(c4);  // 4 consecutive K -> 4 consecutive slots
    unsigned short* dh = hi + dbase;
    unsigned short* dl = lo + dbase;
    const float vv[4] = {f.x, f.y, f.z, f.w};
    #pragma unroll
    for (int e = 0; e < 4; ++e) {
      const unsigned short h = f2bf(vv[e]);
      dh[e] = h;
      dl[e] = f2bf(vv[e] - bf2f(h));
    }
  }
}

// ---- Kernel 0: one-time weight swizzle+convert (4 matrices) -------------
__global__ __launch_bounds__(256) void wprep_kernel(
    const float* __restrict__ Wq, const float* __restrict__ Wk,
    const float* __restrict__ Wv, const float* __restrict__ Wo,
    unsigned short* __restrict__ wswz)
{
  const int m = blockIdx.x;  // 0..3
  const float* W = (m == 0) ? Wq : (m == 1) ? Wk : (m == 2) ? Wv : Wo;
  unsigned short* hi = wswz + m * WMAT_STRIDE;
  unsigned short* lo = hi + WMAT_ELEMS;
  for (int it = 0; it < 64; ++it) {
    const int f = threadIdx.x + 256 * it;      // 0..16383
    const int e = f & 15, lane = (f >> 4) & 31, kc = (f >> 9) & 3, nt = f >> 11;
    const int kk = kc * 32 + (lane >> 4) * 8 + (e & 7) + (e >> 3) * 16;  // B layout: K = h*8+(e&7)+16*(e>>3) per chunk
    const int n  = nt * 16 + (lane & 15);
    const float val = W[kk * F_DIM + n];
    const unsigned short h = f2bf(val);
    hi[f] = h;
    lo[f] = f2bf(val - bf2f(h));
  }
}

// ---- Kernel 1: fused Q/K/V projection ----------------------------------
__global__ __launch_bounds__(256) void qkv_kernel(
    const float* __restrict__ x, const unsigned short* __restrict__ wswz,
    const float* __restrict__ bq, const float* __restrict__ bk, const float* __restrict__ bv,
    float* __restrict__ q, float* __restrict__ k, float* __restrict__ v)
{
  __shared__ __align__(16) unsigned short xhi[128 * A_STRIDE];
  __shared__ __align__(16) unsigned short xlo[128 * A_STRIDE];
  const int tid = threadIdx.x, lane = tid & 31, wave = tid >> 5;
  const int rowBase = blockIdx.x * 128;

  stage_block(x, rowBase, xhi, xlo, tid);
  __syncthreads();

  const int arow = wave * 16 + (lane & 15);
  v16bf Ah[4], Al[4];
  #pragma unroll
  for (int kc = 0; kc < 4; ++kc) {
    Ah[kc] = load_a(xhi, arow, kc, lane);
    Al[kc] = load_a(xlo, arow, kc, lane);
  }

  #pragma unroll 1
  for (int nt = 0; nt < 24; ++nt) {
    const int tn = nt >> 3, ntile = nt & 7;
    const unsigned short* Wz = wswz + tn * WMAT_STRIDE;
    const float* bb;
    float* O;
    if (tn == 0)      { bb = bq; O = q; }
    else if (tn == 1) { bb = bk; O = k; }
    else              { bb = bv; O = v; }

    const v8f c = mm_tile(Wz, ntile, Ah, Al, lane);

    const int n = ntile * 16 + (lane & 15);
    const float bias = bb[n];
    const int row0 = rowBase + wave * 16 + (lane >> 4) * 8;  // D: M = r + 8*(lane>=16)
    #pragma unroll
    for (int rr = 0; rr < 8; ++rr)
      O[(row0 + rr) * F_DIM + n] = c[rr] + bias;
  }
}

// ---- Kernel 2: dilated causal attention (<=9 keys/query) ---------------
__global__ __launch_bounds__(256) void attn_kernel(
    const float* __restrict__ q, const float* __restrict__ k,
    const float* __restrict__ v, float* __restrict__ o)
{
  const int lane = threadIdx.x & 31;                     // lane = d index (D=32)
  const int wid = (blockIdx.x * 256 + threadIdx.x) >> 5; // one wave per (b,h,i)
  const int i = wid & (S_LEN - 1);
  const int bh = wid >> 12;
  const int h = bh & (H_NUM - 1), b = bh >> 2;
  const int base_i = ((b * S_LEN + i) * H_NUM + h) * D_DIM + lane;
  const float qd = q[base_i];

  float sc[9], vr[9];
  #pragma unroll
  for (int t = 0; t < 9; ++t) {
    const int j = i - 4 * t;
    if (j >= 0) {
      const int base_j = ((b * S_LEN + j) * H_NUM + h) * D_DIM + lane;
      const float kd = k[base_j];
      vr[t] = v[base_j];
      float p = qd * kd;
      #pragma unroll
      for (int m = 16; m >= 1; m >>= 1) p += __shfl_xor(p, m, 32);
      sc[t] = p * 0.17677669529663688f;                  // 1/sqrt(32)
    } else {
      sc[t] = -1e9f;                                     // reference NEG masking
      vr[t] = 0.0f;
    }
  }
  float mx = sc[0];
  #pragma unroll
  for (int t = 1; t < 9; ++t) mx = fmaxf(mx, sc[t]);
  float sum = 0.0f, od = 0.0f;
  #pragma unroll
  for (int t = 0; t < 9; ++t) {
    const float p = __expf(sc[t] - mx);
    sum += p;
    od += p * vr[t];
  }
  o[base_i] = od / sum;
}

// ---- Kernel 3: output projection + residual + LayerNorm ----------------
__global__ __launch_bounds__(256) void out_kernel(
    const float* __restrict__ o, const float* __restrict__ x,
    const unsigned short* __restrict__ WoZ, const float* __restrict__ bo,
    const float* __restrict__ gamma, const float* __restrict__ beta,
    float* __restrict__ out)
{
  // bf16 staging planes (73728B) alias the f32 y-buffer (67584B)
  __shared__ __align__(16) unsigned char smem[2 * 128 * A_STRIDE * 2];
  unsigned short* ohi = (unsigned short*)smem;
  unsigned short* olo = ohi + 128 * A_STRIDE;
  float* ybuf = (float*)smem;

  const int tid = threadIdx.x, lane = tid & 31, wave = tid >> 5;
  const int rowBase = blockIdx.x * 128;

  stage_block(o, rowBase, ohi, olo, tid);
  __syncthreads();

  const int arow = wave * 16 + (lane & 15);
  v16bf Ah[4], Al[4];
  #pragma unroll
  for (int kc = 0; kc < 4; ++kc) {
    Ah[kc] = load_a(ohi, arow, kc, lane);
    Al[kc] = load_a(olo, arow, kc, lane);
  }
  __syncthreads();  // A in registers; LDS reused as ybuf below

  #pragma unroll 1
  for (int nt = 0; nt < 8; ++nt) {
    const v8f c = mm_tile(WoZ, nt, Ah, Al, lane);
    const int n = nt * 16 + (lane & 15);
    const float bias = bo[n];
    const int lrow = wave * 16 + (lane >> 4) * 8;
    #pragma unroll
    for (int rr = 0; rr < 8; ++rr)
      ybuf[(lrow + rr) * Y_STRIDE + n] = c[rr] + bias;
  }
  __syncthreads();

  // residual + LayerNorm: each wave handles 16 rows; lane covers 4 cols/row
  #pragma unroll 1
  for (int rr = 0; rr < 16; ++rr) {
    const int row = wave * 16 + rr;
    const int grow = rowBase + row;
    float yv[4], sum = 0.0f, sq = 0.0f;
    #pragma unroll
    for (int kk = 0; kk < 4; ++kk) {
      const int cidx = lane + 32 * kk;
      const float yy = ybuf[row * Y_STRIDE + cidx] + x[grow * F_DIM + cidx];
      yv[kk] = yy;
      sum += yy;
      sq += yy * yy;
    }
    #pragma unroll
    for (int m = 16; m >= 1; m >>= 1) {
      sum += __shfl_xor(sum, m, 32);
      sq  += __shfl_xor(sq, m, 32);
    }
    const float mu = sum * (1.0f / 128.0f);
    const float var = sq * (1.0f / 128.0f) - mu * mu;
    const float rinv = rsqrtf(var + 1e-3f);
    #pragma unroll
    for (int kk = 0; kk < 4; ++kk) {
      const int cidx = lane + 32 * kk;
      out[grow * F_DIM + cidx] = (yv[kk] - mu) * rinv * gamma[cidx] + beta[cidx];
    }
  }
}

extern "C" void kernel_launch(void* const* d_in, const int* in_sizes, int n_in,
                              void* d_out, int out_size, void* d_ws, size_t ws_size,
                              hipStream_t stream) {
  (void)in_sizes; (void)n_in; (void)out_size; (void)ws_size;
  const float* x     = (const float*)d_in[0];
  const float* Wq    = (const float*)d_in[1];
  const float* bq    = (const float*)d_in[2];
  const float* Wk    = (const float*)d_in[3];
  const float* bk    = (const float*)d_in[4];
  const float* Wv    = (const float*)d_in[5];
  const float* bv    = (const float*)d_in[6];
  const float* Wo    = (const float*)d_in[7];
  const float* bo    = (const float*)d_in[8];
  const float* gamma = (const float*)d_in[9];
  const float* beta  = (const float*)d_in[10];
  float* out = (float*)d_out;

  // workspace: q,k,v,o each M_ROWS*F_DIM f32 (8 MB) + 4 swizzled bf16 weight mats
  float* q = (float*)d_ws;
  float* k = q + (size_t)M_ROWS * F_DIM;
  float* v = k + (size_t)M_ROWS * F_DIM;
  float* o = v + (size_t)M_ROWS * F_DIM;
  unsigned short* wswz = (unsigned short*)(o + (size_t)M_ROWS * F_DIM);

  wprep_kernel<<<4, 256, 0, stream>>>(Wq, Wk, Wv, Wo, wswz);
  qkv_kernel<<<M_ROWS / 128, 256, 0, stream>>>(x, wswz, bq, bk, bv, q, k, v);
  attn_kernel<<<(M_ROWS * H_NUM) / 8, 256, 0, stream>>>(q, k, v, o);
  out_kernel<<<M_ROWS / 128, 256, 0, stream>>>(o, x, wswz + 3 * WMAT_STRIDE, bo,
                                               gamma, beta, out);
}
